// SAttnRNNBlock_12833362281081
// MI455X (gfx1250) — matrix operand order repaired and checked by reading, not compiled, not run on hardware
//
#include <hip/hip_runtime.h>

#define DEV static __device__ __forceinline__

typedef __attribute__((ext_vector_type(16))) __bf16 v16bf;
typedef __attribute__((ext_vector_type(8)))  __bf16 v8bf;
typedef __attribute__((ext_vector_type(8)))  float  v8f;

constexpr int B_ = 8, S_ = 2048, D_ = 1024, L_ = 2;
constexpr int M_ = B_ * S_;              // 16384 flattened (b,s) rows
constexpr float SCALE_ = 0.03125f;       // 1/sqrt(1024)
constexpr int RNN_WGS = 64;

// ---------- bf16 helpers (RNE convert) ----------
DEV __bf16 f2bf(float f) {
  unsigned u = __builtin_bit_cast(unsigned, f);
  unsigned r = (u + 0x7FFFu + ((u >> 16) & 1u)) >> 16;
  unsigned short s = (unsigned short)r;
  return __builtin_bit_cast(__bf16, s);
}

// Compiler barrier that also escapes an LDS pointer: the TDM engine writes LDS
// behind the compiler's back (the builtin only sees an integer offset), so
// without this clang proves the __shared__ buffer is never stored and folds
// all loads from it (observed in round-2 asm: B stuck in v[0:7]).
DEV void lds_fence(void* p) { asm volatile("" : "+v"(p) : : "memory"); }

// A-matrix 16x32 bf16 tile, contiguous k order (same as the B loads):
// lane<16: elems 0..15 -> K 0..15 (row = lane&15)
// lane>=16: elems 0..15 -> K 16..31
// caller passes p = rowPtr + kt + ((lane<16)?0:16)
DEV v16bf load_a_tile(const __bf16* p) {
  v8bf c0 = *(const v8bf*)(p);
  v8bf c1 = *(const v8bf*)(p + 8);
  v16bf a;
#pragma unroll
  for (int i = 0; i < 8; ++i) { a[i] = c0[i]; a[i + 8] = c1[i]; }
  return a;
}

DEV v8f wmma_bf16(v16bf a, v16bf b, v8f c) {
  return __builtin_amdgcn_wmma_f32_16x16x32_bf16(false, a, false, b, (short)0, c,
                                                 false, false);
}

// ---------- Tensor Data Mover: 2-D tile load (global -> LDS) ----------
#if __has_builtin(__builtin_amdgcn_tensor_load_to_lds)
#define HAVE_TDM 1
typedef __attribute__((ext_vector_type(4))) unsigned v4u;
typedef __attribute__((ext_vector_type(8))) int      v8i;
typedef __attribute__((ext_vector_type(4))) int      v4i;

// D# per CDNA5 ISA. data_size=1 (2-byte elems).
DEV void tdm_load_2d(const void* gaddr, unsigned lds_off,
                     unsigned tile_d0, unsigned tile_d1,
                     unsigned tens_d0, unsigned tens_d1,
                     unsigned long long stride0_elems) {
  unsigned long long ga = (unsigned long long)gaddr;
  v4u g0 = { 1u,                                   // count=1, is_restore=0
             lds_off,                              // lds_addr [63:32]
             (unsigned)ga,                         // global_addr lo
             (unsigned)((ga >> 32) & 0x1FFFFFFu) | (2u << 30) };  // hi + type=2
  v8i g1;
  g1[0] = (int)(1u << 16);                                       // data_size=2B
  g1[1] = (int)((tens_d0 & 0xFFFFu) << 16);                      // dim0 lo16
  g1[2] = (int)((tens_d0 >> 16) | ((tens_d1 & 0xFFFFu) << 16));  // dim0 hi|dim1 lo
  g1[3] = (int)((tens_d1 >> 16) | (tile_d0 << 16));              // dim1 hi|tile0
  g1[4] = (int)tile_d1;                                          // tile1, tile2=0
  g1[5] = (int)(unsigned)stride0_elems;                          // stride0 lo32
  g1[6] = (int)(unsigned)(stride0_elems >> 32);                  // stride0 hi16
  g1[7] = 0;
  v4i z = {0, 0, 0, 0};
#if __clang_major__ >= 23
  v8i z8 = {};
  __builtin_amdgcn_tensor_load_to_lds(g0, g1, z, z, z8, 0);
#else
  __builtin_amdgcn_tensor_load_to_lds(g0, g1, z, z, 0);
#endif
}
#endif

// ---------- fp32 -> bf16 conversion ----------
__global__ void k_f32_to_bf16(const float* __restrict__ in, __bf16* __restrict__ out,
                              size_t n) {
  size_t i = (size_t)blockIdx.x * blockDim.x + threadIdx.x;
  size_t stride = (size_t)gridDim.x * blockDim.x;
  for (; i < n; i += stride) out[i] = f2bf(in[i]);
}

// ---------- WMMA GEMM: out = A(bf16,MxK) @ W(bf16,NxK)^T + bias(+bias2) ------
// Block: 8 waves, 256-row x 64-col tile. Wave w: rows w*32..w*32+31 (2 m-tiles)
// x 4 n-tiles. B tile (64 rows x 32 k) staged per k-step via TDM, double-
// buffered in LDS, shared by all 8 waves. outBT writes bf16 transposed (B,D,S).
__global__ void k_gemm_bf16(const __bf16* __restrict__ A, const __bf16* __restrict__ W,
                            const float* __restrict__ bias, const float* __restrict__ bias2,
                            float* __restrict__ outF, __bf16* __restrict__ outB,
                            __bf16* __restrict__ outBT,
                            int Mrows, int N, int K) {
  __shared__ __bf16 Bsh[2][64][32];      // single shared alloc -> LDS offset 0
  const int lane = threadIdx.x & 31;
  const int w = threadIdx.x >> 5;
  const int nGroups = N >> 6;
  const int mBlock = blockIdx.x / nGroups;
  const int nGroup = blockIdx.x % nGroups;
  const int r0 = mBlock * 256 + w * 32 + (lane & 15);
  const int aoff = (lane < 16) ? 0 : 16;
  const int kb0  = (lane < 16) ? 0 : 16;
  const __bf16* arow0 = A + (size_t)r0 * K;
  const __bf16* arow1 = arow0 + (size_t)16 * K;
  const __bf16* Wtile = W + (size_t)(nGroup * 64) * K;

  v8f acc[2][4] = {};
#if HAVE_TDM
  if (w == 0) tdm_load_2d(Wtile, 0u, 32, 64, (unsigned)K, 64, (unsigned)K);
  for (int kt = 0; kt < K; kt += 32) {
    const int cur = (kt >> 5) & 1;
    if (w == 0) {
      if (kt + 32 < K) {
        tdm_load_2d(Wtile + kt + 32, (unsigned)(cur ^ 1) * 4096u, 32, 64,
                    (unsigned)K, 64, (unsigned)K);
        __builtin_amdgcn_s_wait_tensorcnt(1);
      } else {
        __builtin_amdgcn_s_wait_tensorcnt(0);
      }
    }
    __syncthreads();                     // Bsh[cur] visible to all waves
    lds_fence(&Bsh[0][0][0]);            // TDM wrote LDS: force real reloads
    v16bf a0 = load_a_tile(arow0 + kt + aoff);
    v16bf a1 = load_a_tile(arow1 + kt + aoff);
#pragma unroll
    for (int t = 0; t < 4; ++t) {
      v16bf b = *(const v16bf*)(&Bsh[cur][t * 16 + (lane & 15)][kb0]);
      acc[0][t] = wmma_bf16(a0, b, acc[0][t]);
      acc[1][t] = wmma_bf16(a1, b, acc[1][t]);
    }
    __syncthreads();                     // reads done before buffer reuse
  }
#else
  for (int kt = 0; kt < K; kt += 32) {
    v16bf a0 = load_a_tile(arow0 + kt + aoff);
    v16bf a1 = load_a_tile(arow1 + kt + aoff);
#pragma unroll
    for (int t = 0; t < 4; ++t) {
      int n = nGroup * 64 + t * 16 + (lane & 15);
      v16bf b = *(const v16bf*)(W + (size_t)n * K + kt + kb0);
      acc[0][t] = wmma_bf16(a0, b, acc[0][t]);
      acc[1][t] = wmma_bf16(a1, b, acc[1][t]);
    }
  }
#endif
#pragma unroll
  for (int mt = 0; mt < 2; ++mt)
#pragma unroll
    for (int t = 0; t < 4; ++t) {
      int n = nGroup * 64 + t * 16 + (lane & 15);
      float bs = (bias ? bias[n] : 0.f) + (bias2 ? bias2[n] : 0.f);
#pragma unroll
      for (int j = 0; j < 8; ++j) {
        int m = j + ((lane < 16) ? 0 : 8);
        int rr = mBlock * 256 + w * 32 + mt * 16 + m;
        float v = acc[mt][t][j] + bs;
        if (outF) outF[(size_t)rr * N + n] = v;
        if (outB) outB[(size_t)rr * N + n] = f2bf(v);
        if (outBT) {                     // transposed (B, N, S_): rr = b*S_ + t
          int bb = rr >> 11, ts = rr & (S_ - 1);   // S_ == 2048
          outBT[((size_t)bb * N + n) * S_ + ts] = f2bf(v);
        }
      }
    }
}

// ---------- persistent RNN recurrence (latency-bound; direct loads) ----------
__global__ void k_rnn(const float* __restrict__ Xg, const __bf16* __restrict__ Whh,
                      __bf16* __restrict__ Hbf, __bf16* __restrict__ hbuf,
                      unsigned* __restrict__ bar) {
  const int lane = threadIdx.x;            // blockDim == 32
  const int n = blockIdx.x * 16 + (lane & 15);
  const int aoff = (lane < 16) ? 0 : 16;
  const int kb0  = (lane < 16) ? 0 : 16;
  const __bf16* brow = Whh + (size_t)n * D_;

  unsigned epoch = 0;
  int sel = 0;
  for (int t = 0; t < S_; ++t) {
    const __bf16* arow = hbuf + (size_t)sel * 16 * D_ + (size_t)(lane & 15) * D_;
    v8f acc = {};
    for (int kt = 0; kt < D_; kt += 32) {
      v16bf a = load_a_tile(arow + kt + aoff);
      v16bf b = *(const v16bf*)(brow + kt + kb0);
      acc = wmma_bf16(a, b, acc);
    }
    if (lane < 16) {                       // rows m=j (j=0..7) are real batches
#pragma unroll
      for (int j = 0; j < 8; ++j) {
        float g = Xg[(size_t)j * S_ * D_ + (size_t)t * D_ + n];
        float h = tanhf(acc[j] + g);
        __bf16 hb = f2bf(h);
        hbuf[(size_t)(sel ^ 1) * 16 * D_ + (size_t)j * D_ + n] = hb;
        Hbf[((size_t)j * S_ + t) * D_ + n] = hb;
      }
    }
    __threadfence();
    if (lane == 0) {
      unsigned prev = atomicAdd(bar, 1u);
      if (prev == RNN_WGS - 1) {
        atomicExch(bar, 0u);
        __threadfence();
        atomicAdd(bar + 1, 1u);            // release epoch t+1
      } else {
        while (atomicAdd(bar + 1, 0u) < epoch + 1) __builtin_amdgcn_s_sleep(2);
      }
    }
    __threadfence();
    ++epoch;
    sel ^= 1;
  }
}

// ---------- flash-style masked attention ----------
// grid (S/16, B), 8 waves. Wave w owns output d-cols [128w,128w+128).
// VT is V transposed (B, D, S): PV B-tiles are contiguous along tokens.
// Per key-step a 1024 x 32-token V^T tile (64KB) is staged via TDM.
struct AttnShared {
  __bf16 Vt[2][D_][32];                  // 128KB (first member -> LDS offset 0)
  float part[8][2][8][32];               // 16KB
  __bf16 Pbuf[16][32];                   // 1KB
};

__global__ void k_attn(const __bf16* __restrict__ Q, const __bf16* __restrict__ K,
                       const __bf16* __restrict__ VT, const unsigned char* __restrict__ mask,
                       __bf16* __restrict__ Out) {
  __shared__ AttnShared sh;
  const int lane = threadIdx.x & 31;
  const int w = threadIdx.x >> 5;
  const int qt = blockIdx.x, b = blockIdx.y;
  const size_t rowbase = (size_t)b * S_ + (size_t)qt * 16;
  const int aoff = (lane < 16) ? 0 : 16;
  const int kb0  = (lane < 16) ? 0 : 16;
  const int dlo = w * 128;
  const int mrow = ((lane < 16) ? 0 : 8);
  const int voff = ((lane < 16) ? 0 : 16);

  const __bf16* qrow = Q + (rowbase + (lane & 15)) * D_;
  v16bf qa[4];
#pragma unroll
  for (int c = 0; c < 4; ++c) qa[c] = load_a_tile(qrow + dlo + c * 32 + aoff);

  const unsigned char* mbase = mask + ((size_t)b * S_ + (size_t)qt * 16) * S_;

  v8f acc[8] = {};
  float mrun[8], lrun[8];
#pragma unroll
  for (int j = 0; j < 8; ++j) { mrun[j] = -1e30f; lrun[j] = 0.f; }

#if HAVE_TDM
  if (w == 0)
    tdm_load_2d(VT + (size_t)b * D_ * S_, 0u, 32, D_, S_, D_, S_);
#endif

  for (int tb = 0; tb < S_; tb += 32) {
    const int cur = (tb >> 5) & 1;
#if HAVE_TDM
    if (w == 0) {
      if (tb + 32 < S_) {
        tdm_load_2d(VT + (size_t)b * D_ * S_ + tb + 32,
                    (unsigned)(cur ^ 1) * (unsigned)sizeof(sh.Vt[0]),
                    32, D_, S_, D_, S_);
        __builtin_amdgcn_s_wait_tensorcnt(1);
      } else {
        __builtin_amdgcn_s_wait_tensorcnt(0);
      }
    }
#endif
    // ---- partial scores over my d-slice, two 16-key tiles ----
    v8f sc[2] = {};
#pragma unroll
    for (int tt = 0; tt < 2; ++tt) {
      const __bf16* krow = K + ((size_t)b * S_ + tb + tt * 16 + (lane & 15)) * D_;
#pragma unroll
      for (int c = 0; c < 4; ++c) {
        v16bf kb = *(const v16bf*)(krow + dlo + c * 32 + kb0);
        sc[tt] = wmma_bf16(qa[c], kb, sc[tt]);
      }
    }
    __syncthreads();                     // publishes Vt[cur] to all waves
    lds_fence(&sh);                      // TDM wrote LDS: force real reloads
#pragma unroll
    for (int tt = 0; tt < 2; ++tt)
#pragma unroll
      for (int j = 0; j < 8; ++j) sh.part[w][tt][j][lane] = sc[tt][j];
    __syncthreads();

    // ---- reduce partials -> full scores; scale + mask ----
    float s[2][8];
#pragma unroll
    for (int tt = 0; tt < 2; ++tt)
#pragma unroll
      for (int j = 0; j < 8; ++j) {
        float v = 0.f;
#pragma unroll
        for (int ww = 0; ww < 8; ++ww) v += sh.part[ww][tt][j][lane];
        v *= SCALE_;
        int m = j + mrow;
        int tk = tb + tt * 16 + (lane & 15);
        if (mbase[(size_t)m * S_ + tk]) v = -1e9f;
        s[tt][j] = v;
      }

    // ---- online softmax (row reductions across 16-lane halves) ----
#pragma unroll
    for (int j = 0; j < 8; ++j) {
      float mx = fmaxf(s[0][j], s[1][j]);
      mx = fmaxf(mx, __shfl_xor(mx, 1, 32));
      mx = fmaxf(mx, __shfl_xor(mx, 2, 32));
      mx = fmaxf(mx, __shfl_xor(mx, 4, 32));
      mx = fmaxf(mx, __shfl_xor(mx, 8, 32));
      float mnew = fmaxf(mrun[j], mx);
      float corr = __expf(mrun[j] - mnew);
      float p0 = __expf(s[0][j] - mnew), p1 = __expf(s[1][j] - mnew);
      float rs = p0 + p1;
      rs += __shfl_xor(rs, 1, 32);
      rs += __shfl_xor(rs, 2, 32);
      rs += __shfl_xor(rs, 4, 32);
      rs += __shfl_xor(rs, 8, 32);
      lrun[j] = lrun[j] * corr + rs;
      mrun[j] = mnew;
      s[0][j] = p0; s[1][j] = p1;
#pragma unroll
      for (int t2 = 0; t2 < 8; ++t2) acc[t2][j] *= corr;
    }
    __syncthreads();
    if (w == 0) {                        // stage P (C-layout regs -> LDS)
#pragma unroll
      for (int tt = 0; tt < 2; ++tt)
#pragma unroll
        for (int j = 0; j < 8; ++j)
          sh.Pbuf[j + mrow][tt * 16 + (lane & 15)] = f2bf(s[tt][j]);
    }
    __syncthreads();

    // ---- P (16x32) x V (32x16 per tile) accumulated into acc ----
    v16bf pa;
    {
      int r = lane & 15;
#pragma unroll
      for (int i = 0; i < 16; ++i) {
        int k = i + ((lane < 16) ? 0 : 16);
        pa[i] = sh.Pbuf[r][k];
      }
    }
#pragma unroll
    for (int t2 = 0; t2 < 8; ++t2) {
      int n = dlo + t2 * 16 + (lane & 15);
#if HAVE_TDM
      v16bf vb = *(const v16bf*)(&sh.Vt[cur][n][voff]);   // contiguous 32B
#else
      v16bf vb = *(const v16bf*)(VT + ((size_t)b * D_ + n) * S_ + tb + voff);
#endif
      acc[t2] = wmma_bf16(pa, vb, acc[t2]);
    }
    __syncthreads();                     // PV reads done before Vt reuse
  }

  float invl[8];
#pragma unroll
  for (int j = 0; j < 8; ++j) invl[j] = 1.0f / lrun[j];
#pragma unroll
  for (int t2 = 0; t2 < 8; ++t2) {
    int n = dlo + t2 * 16 + (lane & 15);
#pragma unroll
    for (int j = 0; j < 8; ++j) {
      int m = j + mrow;
      Out[(rowbase + m) * D_ + n] = f2bf(acc[t2][j] * invl[j]);
    }
  }
}

// ---------------------------------------------------------------------------
extern "C" void kernel_launch(void* const* d_in, const int* in_sizes, int n_in,
                              void* d_out, int out_size, void* d_ws, size_t ws_size,
                              hipStream_t stream) {
  (void)in_sizes; (void)n_in; (void)out_size; (void)ws_size;
  const unsigned char* mask = (const unsigned char*)d_in[0];   // bool (B,S,S)
  const float* reps = (const float*)d_in[1];
  const float* W_ih = (const float*)d_in[2];
  const float* W_hh = (const float*)d_in[3];
  const float* b_ih = (const float*)d_in[4];
  const float* b_hh = (const float*)d_in[5];
  const float* Wq   = (const float*)d_in[6];
  const float* bq   = (const float*)d_in[7];
  const float* Wk   = (const float*)d_in[8];
  const float* bk   = (const float*)d_in[9];
  const float* Wv   = (const float*)d_in[10];
  const float* bv   = (const float*)d_in[11];
  const float* Wo   = (const float*)d_in[12];
  const float* bo   = (const float*)d_in[13];

  const size_t MD = (size_t)M_ * D_;      // 16M elements
  const size_t WW = (size_t)L_ * D_ * D_; // 2M elements per weight family
  size_t off = 0;
  auto carve = [&](size_t bytes) {
    char* p = (char*)d_ws + off;
    off += (bytes + 255) & ~(size_t)255;
    return (void*)p;
  };
  __bf16* Xbf  = (__bf16*)carve(MD * 2);            // layer input (bf16)
  __bf16* Hbf  = (__bf16*)carve(MD * 2);            // RNN hidden / attn out (reused)
  float*  Xg   = (float*) carve(MD * 4);            // gate precompute (f32)
  __bf16* Qbf  = (__bf16*)Xg;                       // alias: Xg dead after RNN
  __bf16* Kbf  = (__bf16*)Xg + MD;
  __bf16* VTbf = (__bf16*)carve(MD * 2);            // V transposed (B, D, S)
  __bf16* Wihb = (__bf16*)carve(WW * 2);
  __bf16* Whhb = (__bf16*)carve(WW * 2);
  __bf16* Wqb  = (__bf16*)carve(WW * 2);
  __bf16* Wkb  = (__bf16*)carve(WW * 2);
  __bf16* Wvb  = (__bf16*)carve(WW * 2);
  __bf16* Wob  = (__bf16*)carve(WW * 2);
  __bf16* hbuf = (__bf16*)carve(2 * 16 * D_ * 2);   // double-buffered padded h
  unsigned* bar = (unsigned*)carve(256);

  k_f32_to_bf16<<<2048, 256, 0, stream>>>(reps, Xbf, MD);
  k_f32_to_bf16<<<1024, 256, 0, stream>>>(W_ih, Wihb, WW);
  k_f32_to_bf16<<<1024, 256, 0, stream>>>(W_hh, Whhb, WW);
  k_f32_to_bf16<<<1024, 256, 0, stream>>>(Wq, Wqb, WW);
  k_f32_to_bf16<<<1024, 256, 0, stream>>>(Wk, Wkb, WW);
  k_f32_to_bf16<<<1024, 256, 0, stream>>>(Wv, Wvb, WW);
  k_f32_to_bf16<<<1024, 256, 0, stream>>>(Wo, Wob, WW);

  const int gemmBlocks = (M_ / 256) * (D_ / 64);     // 1024 blocks x 8 waves

  for (int l = 0; l < L_; ++l) {
    const size_t wo_ = (size_t)l * D_ * D_;
    const size_t bo_ = (size_t)l * D_;
    // 1) Xg = X @ Wih^T + bih + bhh  (f32)
    k_gemm_bf16<<<gemmBlocks, 256, 0, stream>>>(Xbf, Wihb + wo_, b_ih + bo_, b_hh + bo_,
                                                Xg, nullptr, nullptr, M_, D_, D_);
    // 2) recurrence state init
    hipMemsetAsync(hbuf, 0, 2 * 16 * D_ * 2, stream);
    hipMemsetAsync(bar, 0, 256, stream);
    // 3) sequential RNN (persistent, device-wide barrier)
    k_rnn<<<RNN_WGS, 32, 0, stream>>>(Xg, Whhb + wo_, Hbf, hbuf, bar);
    // 4) Q/K/V projections (Q/K alias the now-dead Xg region; V transposed)
    k_gemm_bf16<<<gemmBlocks, 256, 0, stream>>>(Hbf, Wqb + wo_, bq + bo_, nullptr,
                                                nullptr, Qbf, nullptr, M_, D_, D_);
    k_gemm_bf16<<<gemmBlocks, 256, 0, stream>>>(Hbf, Wkb + wo_, bk + bo_, nullptr,
                                                nullptr, Kbf, nullptr, M_, D_, D_);
    k_gemm_bf16<<<gemmBlocks, 256, 0, stream>>>(Hbf, Wvb + wo_, bv + bo_, nullptr,
                                                nullptr, nullptr, VTbf, M_, D_, D_);
    // 5) masked softmax attention -> Hbf (reuse)
    k_attn<<<dim3(S_ / 16, B_), 256, 0, stream>>>(Qbf, Kbf, VTbf, mask, Hbf);
    // 6) output projection
    if (l == 0)
      k_gemm_bf16<<<gemmBlocks, 256, 0, stream>>>(Hbf, Wob + wo_, bo + bo_, nullptr,
                                                  nullptr, Xbf, nullptr, M_, D_, D_);
    else
      k_gemm_bf16<<<gemmBlocks, 256, 0, stream>>>(Hbf, Wob + wo_, bo + bo_, nullptr,
                                                  (float*)d_out, nullptr, nullptr, M_, D_, D_);
  }
}
